// TransformerEncoderBlock_13374528160238
// MI455X (gfx1250) — compile-verified, matrix operand order checked by reading, not run on hardware
//
#include <hip/hip_runtime.h>
#include <hip/hip_bf16.h>
#include <math.h>

// ---------- types ----------
typedef __attribute__((ext_vector_type(16))) __bf16 v16bf;
typedef __attribute__((ext_vector_type(8)))  float  v8f;

#define DEV __device__ __forceinline__

static constexpr int Bc  = 8;
static constexpr int Nc  = 2048;
static constexpr int Dc  = 256;
static constexpr int Hc  = 8;
static constexpr int HDc = 32;
static constexpr int DFFc = 1024;
static constexpr int Mc  = Bc * Nc;      // 16384 tokens

// ---------- WMMA helpers ----------
DEV v8f wmma_bf16(v16bf a, v16bf b, v8f c) {
    return __builtin_amdgcn_wmma_f32_16x16x32_bf16(
        false, a, false, b, (short)0, c, false, false);
}

// Load a 16x32 bf16 tile (row-major, leading dim ld) as an A/B fragment.
// lane L holds row (L&15); element j: j<8 -> K=half*8+j ; j>=8 -> K=16+half*8+(j-8)
DEV v16bf load_frag(const __bf16* base, int ld, int lane) {
    const int row  = lane & 15;
    const int half = lane >> 4;
    const __bf16* p = base + (size_t)row * ld + half * 8;
    union Pack { uint4 u[2]; v16bf v; } pk;
    pk.u[0] = *(const uint4*)(p);
    pk.u[1] = *(const uint4*)(p + 16);
    return pk.v;
}
// C/D fragment mapping: element i of lane L -> M = i + 8*(L>>4), N = L&15.

// ---------- async global -> LDS (ASYNCcnt path) ----------
DEV void async_g2l_b128(unsigned lds_off, const void* gptr) {
    asm volatile("global_load_async_to_lds_b128 %0, %1, off"
                 :: "v"(lds_off), "v"((unsigned long long)(uintptr_t)gptr)
                 : "memory");
}
DEV void wait_async0() { asm volatile("s_wait_asynccnt 0" ::: "memory"); }

// ---------- wave32 reductions ----------
DEV float redmax16(float v) {
    v = fmaxf(v, __shfl_xor(v, 1, 32));
    v = fmaxf(v, __shfl_xor(v, 2, 32));
    v = fmaxf(v, __shfl_xor(v, 4, 32));
    v = fmaxf(v, __shfl_xor(v, 8, 32));
    return v;
}
DEV float redsum16(float v) {
    v += __shfl_xor(v, 1, 32);
    v += __shfl_xor(v, 2, 32);
    v += __shfl_xor(v, 4, 32);
    v += __shfl_xor(v, 8, 32);
    return v;
}
DEV float redsum32(float v) {
    v = redsum16(v);
    v += __shfl_xor(v, 16, 32);
    return v;
}

// ---------- fp32 -> bf16 conversion ----------
__global__ void cvt_bf16_kernel(const float* __restrict__ src,
                                __bf16* __restrict__ dst, int n) {
    int i = blockIdx.x * blockDim.x + threadIdx.x;
    if (i < n) dst[i] = (__bf16)src[i];
}

// =====================================================================
// Staged-weight GEMM core (block = 256 thr / 8 waves, tile 128M x 64N).
// W panel [64 x 32] per k-stage staged to LDS with async DMA, double
// buffered; A fragment register-prefetched.  Each kernel instantiates
// this inline body with its own epilogue.
// =====================================================================

// ---------- QKV GEMM: [M,256] @ [768,256]^T, scatter epilogue ----------
__global__ void gemm_qkv_kernel(const __bf16* __restrict__ A,
                                const __bf16* __restrict__ W,
                                __bf16* __restrict__ q,
                                __bf16* __restrict__ k,
                                __bf16* __restrict__ vt,
                                float qscale) {
    __shared__ alignas(16) __bf16 wt[2][64 * 32];
    const int tid  = threadIdx.x;
    const int lane = tid & 31;
    const int wid  = tid >> 5;
    const int row0 = (blockIdx.x * 8 + wid) * 16;
    const int col0 = blockIdx.y * 64;
    const int half = lane >> 4;

    const int srow = tid >> 2, schunk = tid & 3;          // 64 rows x 4 chunks
    const __bf16* gw = W + (size_t)(col0 + srow) * Dc + schunk * 8;
    const unsigned lsub = (unsigned)(srow * 32 + schunk * 8);
    auto issue = [&](int k0, int buf) {
        async_g2l_b128((unsigned)(uintptr_t)&wt[buf][lsub], gw + k0);
    };
    issue(0, 0);

    const __bf16* Ap = A + (size_t)row0 * Dc;
    v8f acc[4] = {};
    v16bf a_cur = load_frag(Ap, Dc, lane);

    constexpr int S = Dc / 32;   // 8 k-stages
    for (int s = 0; s < S; ++s) {
        wait_async0();
        __syncthreads();
        if (s + 1 < S) issue((s + 1) * 32, (s + 1) & 1);
        v16bf a_n = a_cur;
        if (s + 1 < S) a_n = load_frag(Ap + (s + 1) * 32, Dc, lane);
        const __bf16* lb = &wt[s & 1][0];
#pragma unroll
        for (int t = 0; t < 4; ++t) {
            v16bf b = load_frag(lb + t * 16 * 32, 32, lane);
            acc[t] = wmma_bf16(a_cur, b, acc[t]);
        }
        a_cur = a_n;
    }
#pragma unroll
    for (int t = 0; t < 4; ++t) {
#pragma unroll
        for (int i = 0; i < 8; ++i) {
            int m = row0 + i + 8 * half;           // token
            int n = col0 + t * 16 + (lane & 15);   // [0,768)
            float val = acc[t][i];
            int part = n >> 8;                     // 0=q 1=k 2=v
            int hn = n & 255;
            int hh = hn >> 5, d = hn & 31;
            int bb = m >> 11, ii = m & (Nc - 1);
            size_t bh = (size_t)(bb * Hc + hh);
            if (part == 0)      q[(bh * Nc + ii) * HDc + d] = (__bf16)(val * qscale);
            else if (part == 1) k[(bh * Nc + ii) * HDc + d] = (__bf16)val;
            else                vt[(bh * HDc + d) * Nc + ii] = (__bf16)val;  // V^T
        }
    }
}

// ---------- flash attention: grid (B*H, N/64), block 128 (4 waves) ----------
// Scores pre-scaled by HD^-0.5 * log2(e): softmax in exp2 domain,
// exp2f -> single v_exp_f32 (co-executes with XDL WMMA).
__global__ void attn_kernel(const __bf16* __restrict__ q,
                            const __bf16* __restrict__ k,
                            const __bf16* __restrict__ vt,
                            __bf16* __restrict__ ob) {
    __shared__ alignas(16) __bf16 pl[4][16 * 64];
    const int lane = threadIdx.x & 31;
    const int wid  = threadIdx.x >> 5;
    const int half = lane >> 4;
    const int bh = blockIdx.x;
    const int i0 = (blockIdx.y * 4 + wid) * 16;

    const __bf16* qp = q  + ((size_t)bh * Nc + i0) * HDc;
    const __bf16* kp = k  + (size_t)bh * Nc * HDc;
    const __bf16* vp = vt + (size_t)bh * HDc * Nc;

    v16bf qa = load_frag(qp, HDc, lane);   // whole 16 x HD A-fragment

    v8f acc0 = {}, acc1 = {};              // O tile: d 0..15 / 16..31
    float mrow[8], lrow[8];
#pragma unroll
    for (int i = 0; i < 8; ++i) { mrow[i] = -1e30f; lrow[i] = 0.f; }

    for (int jc = 0; jc < Nc; jc += 64) {
        v8f z = {};
        v8f s0 = wmma_bf16(qa, load_frag(kp + (size_t)(jc     ) * HDc, HDc, lane), z);
        v8f s1 = wmma_bf16(qa, load_frag(kp + (size_t)(jc + 16) * HDc, HDc, lane), z);
        v8f s2 = wmma_bf16(qa, load_frag(kp + (size_t)(jc + 32) * HDc, HDc, lane), z);
        v8f s3 = wmma_bf16(qa, load_frag(kp + (size_t)(jc + 48) * HDc, HDc, lane), z);

#pragma unroll
        for (int i = 0; i < 8; ++i) {
            float cm = fmaxf(fmaxf(s0[i], s1[i]), fmaxf(s2[i], s3[i]));
            cm = redmax16(cm);
            float mn = fmaxf(mrow[i], cm);
            float al = exp2f(mrow[i] - mn);
            float p0 = exp2f(s0[i] - mn);
            float p1 = exp2f(s1[i] - mn);
            float p2 = exp2f(s2[i] - mn);
            float p3 = exp2f(s3[i] - mn);
            lrow[i] = lrow[i] * al + redsum16(p0 + p1 + p2 + p3);
            mrow[i] = mn;
            acc0[i] *= al;
            acc1[i] *= al;
            int r = i + 8 * half, c = lane & 15;
            pl[wid][r * 64 + c]      = (__bf16)p0;
            pl[wid][r * 64 + 16 + c] = (__bf16)p1;
            pl[wid][r * 64 + 32 + c] = (__bf16)p2;
            pl[wid][r * 64 + 48 + c] = (__bf16)p3;
        }
        asm volatile("s_wait_dscnt 0" ::: "memory");  // same-wave LDS WAR/RAW

        v16bf pa0 = load_frag(&pl[wid][0],  64, lane);   // P cols 0..31
        v16bf pa1 = load_frag(&pl[wid][32], 64, lane);   // P cols 32..63
        v16bf vb00 = load_frag(vp + jc,                    Nc, lane);
        v16bf vb01 = load_frag(vp + jc + 32,               Nc, lane);
        v16bf vb10 = load_frag(vp + (size_t)16 * Nc + jc,      Nc, lane);
        v16bf vb11 = load_frag(vp + (size_t)16 * Nc + jc + 32, Nc, lane);
        acc0 = wmma_bf16(pa0, vb00, acc0);
        acc0 = wmma_bf16(pa1, vb01, acc0);
        acc1 = wmma_bf16(pa0, vb10, acc1);
        acc1 = wmma_bf16(pa1, vb11, acc1);
    }

    const int b = bh >> 3, h = bh & 7;
#pragma unroll
    for (int i = 0; i < 8; ++i) {
        float inv = 1.f / lrow[i];
        int m = b * Nc + i0 + i + 8 * half;
        int dcol = lane & 15;
        ob[(size_t)m * Dc + h * HDc + dcol]      = (__bf16)(acc0[i] * inv);
        ob[(size_t)m * Dc + h * HDc + 16 + dcol] = (__bf16)(acc1[i] * inv);
    }
}

// ---------- out-proj: y = o @ w_out^T + b_out (staged weights) ----------
__global__ void gemm_out_kernel(const __bf16* __restrict__ A,
                                const __bf16* __restrict__ W,
                                const float* __restrict__ bias,
                                float* __restrict__ yf,
                                __bf16* __restrict__ yb) {
    __shared__ alignas(16) __bf16 wt[2][64 * 32];
    const int tid  = threadIdx.x;
    const int lane = tid & 31;
    const int wid  = tid >> 5;
    const int row0 = (blockIdx.x * 8 + wid) * 16;
    const int col0 = blockIdx.y * 64;
    const int half = lane >> 4;

    const int srow = tid >> 2, schunk = tid & 3;
    const __bf16* gw = W + (size_t)(col0 + srow) * Dc + schunk * 8;
    const unsigned lsub = (unsigned)(srow * 32 + schunk * 8);
    auto issue = [&](int k0, int buf) {
        async_g2l_b128((unsigned)(uintptr_t)&wt[buf][lsub], gw + k0);
    };
    issue(0, 0);

    const __bf16* Ap = A + (size_t)row0 * Dc;
    v8f acc[4] = {};
    v16bf a_cur = load_frag(Ap, Dc, lane);

    constexpr int S = Dc / 32;
    for (int s = 0; s < S; ++s) {
        wait_async0();
        __syncthreads();
        if (s + 1 < S) issue((s + 1) * 32, (s + 1) & 1);
        v16bf a_n = a_cur;
        if (s + 1 < S) a_n = load_frag(Ap + (s + 1) * 32, Dc, lane);
        const __bf16* lb = &wt[s & 1][0];
#pragma unroll
        for (int t = 0; t < 4; ++t) {
            v16bf b = load_frag(lb + t * 16 * 32, 32, lane);
            acc[t] = wmma_bf16(a_cur, b, acc[t]);
        }
        a_cur = a_n;
    }
#pragma unroll
    for (int t = 0; t < 4; ++t)
#pragma unroll
        for (int i = 0; i < 8; ++i) {
            int m = row0 + i + 8 * half;
            int n = col0 + t * 16 + (lane & 15);
            float val = acc[t][i] + bias[n];
            yf[(size_t)m * Dc + n] = val;
            yb[(size_t)m * Dc + n] = (__bf16)val;
        }
}

// ---------- GEGLU GEMM (both weight panels staged) ----------
__global__ void gemm_geglu_kernel(const __bf16* __restrict__ A,
                                  const __bf16* __restrict__ W1,
                                  const __bf16* __restrict__ Wv,
                                  __bf16* __restrict__ hb) {
    __shared__ alignas(16) __bf16 wt1[2][64 * 32];
    __shared__ alignas(16) __bf16 wtv[2][64 * 32];
    const int tid  = threadIdx.x;
    const int lane = tid & 31;
    const int wid  = tid >> 5;
    const int row0 = (blockIdx.x * 8 + wid) * 16;
    const int col0 = blockIdx.y * 64;
    const int half = lane >> 4;

    const int srow = tid >> 2, schunk = tid & 3;
    const __bf16* g1 = W1 + (size_t)(col0 + srow) * Dc + schunk * 8;
    const __bf16* gv = Wv + (size_t)(col0 + srow) * Dc + schunk * 8;
    const unsigned lsub = (unsigned)(srow * 32 + schunk * 8);
    auto issue = [&](int k0, int buf) {
        async_g2l_b128((unsigned)(uintptr_t)&wt1[buf][lsub], g1 + k0);
        async_g2l_b128((unsigned)(uintptr_t)&wtv[buf][lsub], gv + k0);
    };
    issue(0, 0);

    const __bf16* Ap = A + (size_t)row0 * Dc;
    v8f acc[4] = {}, accv[4] = {};
    v16bf a_cur = load_frag(Ap, Dc, lane);

    constexpr int S = Dc / 32;
    for (int s = 0; s < S; ++s) {
        wait_async0();
        __syncthreads();
        if (s + 1 < S) issue((s + 1) * 32, (s + 1) & 1);
        v16bf a_n = a_cur;
        if (s + 1 < S) a_n = load_frag(Ap + (s + 1) * 32, Dc, lane);
        const __bf16* l1 = &wt1[s & 1][0];
        const __bf16* lv = &wtv[s & 1][0];
#pragma unroll
        for (int t = 0; t < 4; ++t) {
            v16bf b1 = load_frag(l1 + t * 16 * 32, 32, lane);
            v16bf bv = load_frag(lv + t * 16 * 32, 32, lane);
            acc[t]  = wmma_bf16(a_cur, b1, acc[t]);
            accv[t] = wmma_bf16(a_cur, bv, accv[t]);
        }
        a_cur = a_n;
    }
#pragma unroll
    for (int t = 0; t < 4; ++t)
#pragma unroll
        for (int i = 0; i < 8; ++i) {
            int m = row0 + i + 8 * half;
            int n = col0 + t * 16 + (lane & 15);
            float u = acc[t][i];
            float g = 0.5f * u * (1.0f + erff(u * 0.70710678118654752f));
            hb[(size_t)m * DFFc + n] = (__bf16)(g * accv[t][i]);
        }
}

// ---------- FFN2 (K=1024, staged weights): x1 += h @ w2^T ----------
__global__ void gemm_ffn2_kernel(const __bf16* __restrict__ A,
                                 const __bf16* __restrict__ W,
                                 float* __restrict__ x1) {
    __shared__ alignas(16) __bf16 wt[2][64 * 32];
    const int tid  = threadIdx.x;
    const int lane = tid & 31;
    const int wid  = tid >> 5;
    const int row0 = (blockIdx.x * 8 + wid) * 16;
    const int col0 = blockIdx.y * 64;
    const int half = lane >> 4;

    const int srow = tid >> 2, schunk = tid & 3;
    const __bf16* gw = W + (size_t)(col0 + srow) * DFFc + schunk * 8;
    const unsigned lsub = (unsigned)(srow * 32 + schunk * 8);
    auto issue = [&](int k0, int buf) {
        async_g2l_b128((unsigned)(uintptr_t)&wt[buf][lsub], gw + k0);
    };
    issue(0, 0);

    const __bf16* Ap = A + (size_t)row0 * DFFc;
    v8f acc[4] = {};
    v16bf a_cur = load_frag(Ap, DFFc, lane);

    constexpr int S = DFFc / 32;  // 32 k-stages
    for (int s = 0; s < S; ++s) {
        wait_async0();
        __syncthreads();
        if (s + 1 < S) issue((s + 1) * 32, (s + 1) & 1);
        v16bf a_n = a_cur;
        if (s + 1 < S) a_n = load_frag(Ap + (s + 1) * 32, DFFc, lane);
        const __bf16* lb = &wt[s & 1][0];
#pragma unroll
        for (int t = 0; t < 4; ++t) {
            v16bf b = load_frag(lb + t * 16 * 32, 32, lane);
            acc[t] = wmma_bf16(a_cur, b, acc[t]);
        }
        a_cur = a_n;
    }
#pragma unroll
    for (int t = 0; t < 4; ++t)
#pragma unroll
        for (int i = 0; i < 8; ++i) {
            int m = row0 + i + 8 * half;
            int n = col0 + t * 16 + (lane & 15);
            size_t idx = (size_t)m * Dc + n;
            x1[idx] = x1[idx] + acc[t][i];          // ff + x1, in place
        }
}

// ---------- LayerNorm (wave per row, D=256); res may be null ----------
__global__ void layernorm_kernel(const float* __restrict__ src,
                                 const float* __restrict__ res,
                                 const float* __restrict__ g,
                                 const float* __restrict__ b,
                                 float* __restrict__ out) {
    const int lane = threadIdx.x & 31;
    const int wid  = threadIdx.x >> 5;
    const int row  = blockIdx.x * 8 + wid;
    const size_t base = (size_t)row * Dc;

    float v[8];
    float s = 0.f;
#pragma unroll
    for (int c8 = 0; c8 < 8; ++c8) {
        int c = c8 * 32 + lane;
        float t = src[base + c];
        if (res) t += res[base + c];
        v[c8] = t;
        s += t;
    }
    float mean = redsum32(s) * (1.0f / Dc);
    float var = 0.f;
#pragma unroll
    for (int c8 = 0; c8 < 8; ++c8) {
        float d = v[c8] - mean;
        var += d * d;
    }
    var = redsum32(var) * (1.0f / Dc);
    float inv = rsqrtf(var + 1e-5f);
#pragma unroll
    for (int c8 = 0; c8 < 8; ++c8) {
        int c = c8 * 32 + lane;
        out[base + c] = (v[c8] - mean) * inv * g[c] + b[c];
    }
}

// ---------- host launcher ----------
extern "C" void kernel_launch(void* const* d_in, const int* in_sizes, int n_in,
                              void* d_out, int out_size, void* d_ws, size_t ws_size,
                              hipStream_t stream) {
    (void)in_sizes; (void)n_in; (void)out_size; (void)ws_size;
    const float* x     = (const float*)d_in[0];
    const float* w_qkv = (const float*)d_in[1];
    const float* w_out = (const float*)d_in[2];
    const float* b_out = (const float*)d_in[3];
    const float* w1    = (const float*)d_in[4];
    const float* wv    = (const float*)d_in[5];
    const float* w2    = (const float*)d_in[6];
    const float* ln1_g = (const float*)d_in[7];
    const float* ln1_b = (const float*)d_in[8];
    const float* ln2_g = (const float*)d_in[9];
    const float* ln2_b = (const float*)d_in[10];

    char* p = (char*)d_ws;
    size_t off = 0;
    auto take = [&](size_t bytes) -> char* {
        char* r = p + off;
        off += (bytes + 255) & ~(size_t)255;
        return r;
    };
    __bf16* xb    = (__bf16*)take((size_t)Mc * Dc * 2);
    __bf16* wqkvb = (__bf16*)take((size_t)3 * Hc * HDc * Dc * 2);
    __bf16* woutb = (__bf16*)take((size_t)Dc * Dc * 2);
    __bf16* w1b   = (__bf16*)take((size_t)DFFc * Dc * 2);
    __bf16* wvb   = (__bf16*)take((size_t)DFFc * Dc * 2);
    __bf16* w2b   = (__bf16*)take((size_t)Dc * DFFc * 2);
    __bf16* qb    = (__bf16*)take((size_t)Mc * Dc * 2);     // [b,h,i,d]
    __bf16* kb    = (__bf16*)take((size_t)Mc * Dc * 2);     // [b,h,j,d]
    __bf16* vtb   = (__bf16*)take((size_t)Mc * Dc * 2);     // [b,h,d,j]
    __bf16* obuf  = (__bf16*)take((size_t)Mc * Dc * 2);     // [m, h*HD+d]
    __bf16* hb    = qb;   // reuse q/k/vt/o region (4 x 8MB contiguous = 32MB)
    float*  yf    = (float*)take((size_t)Mc * Dc * 4);
    __bf16* yb    = (__bf16*)take((size_t)Mc * Dc * 2);
    float*  x1    = (float*)take((size_t)Mc * Dc * 4);

    auto cvt = [&](const float* s, __bf16* d, int n) {
        cvt_bf16_kernel<<<(n + 255) / 256, 256, 0, stream>>>(s, d, n);
    };
    cvt(x,     xb,    Mc * Dc);
    cvt(w_qkv, wqkvb, 3 * Hc * HDc * Dc);
    cvt(w_out, woutb, Dc * Dc);
    cvt(w1,    w1b,   DFFc * Dc);
    cvt(wv,    wvb,   DFFc * Dc);
    cvt(w2,    w2b,   Dc * DFFc);

    // HD^-0.5 * log2(e): softmax computed in exp2 domain
    const float qscale = 0.17677669529663689f * 1.4426950408889634f;
    gemm_qkv_kernel<<<dim3(Mc / 128, (3 * Hc * HDc) / 64), 256, 0, stream>>>(
        xb, wqkvb, qb, kb, vtb, qscale);

    attn_kernel<<<dim3(Bc * Hc, Nc / 64), 128, 0, stream>>>(qb, kb, vtb, obuf);

    gemm_out_kernel<<<dim3(Mc / 128, Dc / 64), 256, 0, stream>>>(
        obuf, woutb, b_out, yf, yb);

    layernorm_kernel<<<Mc / 8, 256, 0, stream>>>(yf, x, ln1_g, ln1_b, x1);

    gemm_geglu_kernel<<<dim3(Mc / 128, DFFc / 64), 256, 0, stream>>>(
        yb, w1b, wvb, hb);

    gemm_ffn2_kernel<<<dim3(Mc / 128, Dc / 64), 256, 0, stream>>>(hb, w2b, x1);

    layernorm_kernel<<<Mc / 8, 256, 0, stream>>>(x1, nullptr, ln2_g, ln2_b,
                                                 (float*)d_out);
}
